// MixtureOfExperts_17875653886564
// MI455X (gfx1250) — matrix operand order repaired and checked by reading, not compiled, not run on hardware
//
#include <hip/hip_runtime.h>
#include <hip/hip_bf16.h>

typedef __bf16 bf16_t;
typedef __attribute__((ext_vector_type(16))) __bf16 bfx16;
typedef __attribute__((ext_vector_type(8)))  __bf16 bfx8;
typedef __attribute__((ext_vector_type(2)))  __bf16 bfx2;
typedef __attribute__((ext_vector_type(8)))  float  fx8;

#define EMBED 1024
#define INTER 2048
#define NEXP  8
#define TOPK  2
#define TTOT  8192   // 4 * 2048 tokens

// ---- CDNA5 feature probes -------------------------------------------------
#if defined(__gfx1250__) && __has_builtin(__builtin_amdgcn_global_load_async_to_lds_b128)
#define MOE_ASYNC_LDS 1
typedef int v4i __attribute__((vector_size(16)));
typedef __attribute__((address_space(1))) v4i* gv4i_p;   // global
typedef __attribute__((address_space(3))) v4i* lv4i_p;   // LDS
#define ASYNC_CP16(dst, src) \
  __builtin_amdgcn_global_load_async_to_lds_b128((gv4i_p)(src), (lv4i_p)(dst), 0, 0)
static __device__ __forceinline__ void async_wait0() {
#if __has_builtin(__builtin_amdgcn_s_wait_asynccnt)
  __builtin_amdgcn_s_wait_asynccnt(0);
#else
  asm volatile("s_wait_asynccnt 0x0" ::: "memory");
#endif
}
#else
#define MOE_ASYNC_LDS 0
#endif

static __device__ __forceinline__ bfx16 cat8(bfx8 lo, bfx8 hi) {
  return __builtin_shufflevector(lo, hi, 0,1,2,3,4,5,6,7,8,9,10,11,12,13,14,15);
}
static __device__ __forceinline__ int imin(int a, int b) { return a < b ? a : b; }

// pack two f32 -> packed bf16 dword (k, k+1 pair): exactly one v_cvt_pk_bf16_f32
#if defined(__gfx1250__)
static __device__ __forceinline__ unsigned pack2(float a, float b) {
  unsigned r;
  asm("v_cvt_pk_bf16_f32 %0, %1, %2" : "=v"(r) : "v"(a), "v"(b));
  return r;
}
#else
static __device__ __forceinline__ unsigned pack2(float a, float b) {
  union { __bf16 h[2]; unsigned u; } p;
  p.h[0] = (__bf16)a; p.h[1] = (__bf16)b;
  return p.u;
}
#endif

// ---------------------------------------------------------------------------
// init: zero counters
// ---------------------------------------------------------------------------
__global__ void moe_init_kernel(int* counts, float* Psum) {
  int tid = threadIdx.x;
  if (tid < NEXP) { counts[tid] = 0; Psum[tid] = 0.0f; }
}

// ---------------------------------------------------------------------------
// router: logits -> top2 softmax weights, per-expert gather lists, aux stats
// ---------------------------------------------------------------------------
__global__ __launch_bounds__(256)
void moe_router_kernel(const float* __restrict__ x,
                       const float* __restrict__ gw,
                       int* __restrict__ counts,
                       int* __restrict__ idxArr,
                       float* __restrict__ wgtArr,
                       float* __restrict__ Psum) {
  __shared__ float gwl[NEXP][EMBED];   // 32 KB
  __shared__ float psh[NEXP];
  int tid = threadIdx.x;
  for (int i = tid; i < NEXP * EMBED; i += 256)
    gwl[i >> 10][i & (EMBED - 1)] = gw[i];
  if (tid < NEXP) psh[tid] = 0.0f;
  __syncthreads();

  int t = blockIdx.x * 256 + tid;
  const float* xr = x + (size_t)t * EMBED;
  float l[NEXP];
#pragma unroll
  for (int e = 0; e < NEXP; ++e) l[e] = 0.0f;
  for (int d = 0; d < EMBED; d += 4) {
    float4 xv = *(const float4*)(xr + d);
#pragma unroll
    for (int e = 0; e < NEXP; ++e)
      l[e] += xv.x * gwl[e][d] + xv.y * gwl[e][d + 1] +
              xv.z * gwl[e][d + 2] + xv.w * gwl[e][d + 3];
  }
  // top-2 (first index wins ties, like lax.top_k)
  int e1 = 0; float v1 = l[0];
#pragma unroll
  for (int e = 1; e < NEXP; ++e) if (l[e] > v1) { v1 = l[e]; e1 = e; }
  int e2 = (e1 == 0) ? 1 : 0; float v2 = l[e2];
#pragma unroll
  for (int e = 0; e < NEXP; ++e)
    if (e != e1 && l[e] > v2) { v2 = l[e]; e2 = e; }

  float ex = __expf(v2 - v1);
  float w1 = __builtin_amdgcn_rcpf(1.0f + ex);
  float w2 = ex * w1;

  // full softmax over 8 for P = softmax(logits).mean(0)
  float s = 0.0f; float p[NEXP];
#pragma unroll
  for (int e = 0; e < NEXP; ++e) { p[e] = __expf(l[e] - v1); s += p[e]; }
  float inv = __builtin_amdgcn_rcpf(s);
#pragma unroll
  for (int e = 0; e < NEXP; ++e) atomicAdd(&psh[e], p[e] * inv);

  int pos1 = atomicAdd(&counts[e1], 1);
  idxArr[e1 * TTOT + pos1] = t; wgtArr[e1 * TTOT + pos1] = w1;
  int pos2 = atomicAdd(&counts[e2], 1);
  idxArr[e2 * TTOT + pos2] = t; wgtArr[e2 * TTOT + pos2] = w2;

  __syncthreads();
  if (tid < NEXP) atomicAdd(&Psum[tid], psh[tid]);
}

// ---------------------------------------------------------------------------
// cast: x f32 -> bf16 (done once; A operand for all 9 GEMM passes)
// ---------------------------------------------------------------------------
__global__ __launch_bounds__(256)
void moe_cast_kernel(const float* __restrict__ x, bf16_t* __restrict__ xb) {
  size_t base = ((size_t)blockIdx.x * 256 + threadIdx.x) * 8;
  float4 f0 = *(const float4*)(x + base);
  float4 f1 = *(const float4*)(x + base + 4);
  unsigned o[4];
  o[0] = pack2(f0.x, f0.y); o[1] = pack2(f0.z, f0.w);
  o[2] = pack2(f1.x, f1.y); o[3] = pack2(f1.z, f1.w);
  *(uint4*)(xb + base) = *(uint4*)o;
}

// ---------------------------------------------------------------------------
// fused gate+up GEMM + SwiGLU, double-buffered LDS, async A-tile copy.
// C tile 128x64, 8 waves (2m x 4n); wave: 4 m-subtiles, gate+up accumulators
// share A fragments. K = EMBED, N = INTER. idx==nullptr -> identity rows.
// ---------------------------------------------------------------------------
__global__ __launch_bounds__(256)
void moe_gateup_kernel(const bf16_t* __restrict__ xb,
                       const float* __restrict__ Wg,
                       const float* __restrict__ Wu,
                       bf16_t* __restrict__ h,
                       const int* __restrict__ idx,
                       const int* __restrict__ countPtr) {
  const int K = EMBED, N = INTER;
  __shared__ bf16_t As[2][128][40];   // 80B row stride: 16B aligned, conflict-free
  __shared__ bf16_t Bg[2][64][40];    // transposed [n][k]
  __shared__ bf16_t Bu[2][64][40];
  __shared__ int rows[128];

  int count = countPtr ? *countPtr : TTOT;      // uniform scalar
  int m0 = blockIdx.y * 128;
  if (m0 >= count) return;                       // block-uniform exit
  int n0 = blockIdx.x * 64;
  int tid = threadIdx.x;

  if (tid < 128) {
    int r = m0 + tid;
    rows[tid] = idx ? idx[imin(r, count - 1)] : r;
  }
  __syncthreads();

  int lane = tid & 31, wid = tid >> 5;
  int wm = wid >> 2, wn = wid & 3;
  int lm = lane & 15;
  int lk8  = (lane >> 4) * 8;   // A fragment K sub-offset
  int lk16 = (lane >> 4) * 16;  // unused: B fragment also uses lk8
  int bnrow = wn * 16 + lm;

  fx8 accG[4], accU[4];
#pragma unroll
  for (int i = 0; i < 4; ++i) { accG[i] = (fx8)0.0f; accU[i] = (fx8)0.0f; }

  // A loader: row, k-half (16 bf16 = 32B per thread)
  int ar = tid >> 1, ah = (tid & 1) * 16;
  // B loader: (k,k+1) pair at 4 consecutive n -> packed b32 LDS stores
  int bn4 = (tid & 15) * 4;
  int bk2 = (tid >> 4) * 2;

  const bf16_t* aptr = xb + (size_t)rows[ar] * K + ah;
  const float*  gptr = Wg + (size_t)bk2 * N + n0 + bn4;
  const float*  uptr = Wu + (size_t)bk2 * N + n0 + bn4;

#if !MOE_ASYNC_LDS
  uint4 pa0, pa1;
#endif
  float4 fg0, fg1, fu0, fu1;

  // prologue: fetch tile 0
#if MOE_ASYNC_LDS
  ASYNC_CP16(&As[0][ar][ah],     aptr);
  ASYNC_CP16(&As[0][ar][ah + 8], aptr + 8);
#else
  { const uint4* s = (const uint4*)aptr; pa0 = s[0]; pa1 = s[1]; }
#endif
  fg0 = *(const float4*)gptr; fg1 = *(const float4*)(gptr + N);
  fu0 = *(const float4*)uptr; fu1 = *(const float4*)(uptr + N);
#if !MOE_ASYNC_LDS
  *(uint4*)&As[0][ar][ah]     = pa0;
  *(uint4*)&As[0][ar][ah + 8] = pa1;
#endif
  *(unsigned*)&Bg[0][bn4 + 0][bk2] = pack2(fg0.x, fg1.x);
  *(unsigned*)&Bg[0][bn4 + 1][bk2] = pack2(fg0.y, fg1.y);
  *(unsigned*)&Bg[0][bn4 + 2][bk2] = pack2(fg0.z, fg1.z);
  *(unsigned*)&Bg[0][bn4 + 3][bk2] = pack2(fg0.w, fg1.w);
  *(unsigned*)&Bu[0][bn4 + 0][bk2] = pack2(fu0.x, fu1.x);
  *(unsigned*)&Bu[0][bn4 + 1][bk2] = pack2(fu0.y, fu1.y);
  *(unsigned*)&Bu[0][bn4 + 2][bk2] = pack2(fu0.z, fu1.z);
  *(unsigned*)&Bu[0][bn4 + 3][bk2] = pack2(fu0.w, fu1.w);
#if MOE_ASYNC_LDS
  async_wait0();
#endif
  __syncthreads();

  const int NK = K / 32;
  for (int it = 0; it < NK; ++it) {
    int p = it & 1;
    bool more = (it + 1) < NK;
    if (more) {  // issue next tile's fetches before the WMMA block
      const bf16_t* ap = aptr + (it + 1) * 32;
      const float*  gp = gptr + (size_t)(it + 1) * 32 * N;
      const float*  up = uptr + (size_t)(it + 1) * 32 * N;
      __builtin_prefetch(gp + (size_t)32 * N, 0, 0);
      __builtin_prefetch(up + (size_t)32 * N, 0, 0);
#if MOE_ASYNC_LDS
      int q = p ^ 1;
      ASYNC_CP16(&As[q][ar][ah],     ap);
      ASYNC_CP16(&As[q][ar][ah + 8], ap + 8);
#else
      { const uint4* s = (const uint4*)ap; pa0 = s[0]; pa1 = s[1]; }
#endif
      fg0 = *(const float4*)gp; fg1 = *(const float4*)(gp + N);
      fu0 = *(const float4*)up; fu1 = *(const float4*)(up + N);
    }

    bfx16 bg = cat8(*(const bfx8*)&Bg[p][bnrow][lk8],
                    *(const bfx8*)&Bg[p][bnrow][16 + lk8]);
    bfx16 bu = cat8(*(const bfx8*)&Bu[p][bnrow][lk8],
                    *(const bfx8*)&Bu[p][bnrow][16 + lk8]);
    bfx16 afrag[4];
#pragma unroll
    for (int i = 0; i < 4; ++i) {
      int r = wm * 64 + i * 16 + lm;
      afrag[i] = cat8(*(const bfx8*)&As[p][r][lk8],
                      *(const bfx8*)&As[p][r][16 + lk8]);
    }
#pragma unroll
    for (int i = 0; i < 4; ++i) {
      accG[i] = __builtin_amdgcn_wmma_f32_16x16x32_bf16(false, afrag[i], false, bg,
                                                        (short)0, accG[i], false, false);
      accU[i] = __builtin_amdgcn_wmma_f32_16x16x32_bf16(false, afrag[i], false, bu,
                                                        (short)0, accU[i], false, false);
    }

    if (more) {
      int q = p ^ 1;
#if !MOE_ASYNC_LDS
      *(uint4*)&As[q][ar][ah]     = pa0;
      *(uint4*)&As[q][ar][ah + 8] = pa1;
#endif
      *(unsigned*)&Bg[q][bn4 + 0][bk2] = pack2(fg0.x, fg1.x);
      *(unsigned*)&Bg[q][bn4 + 1][bk2] = pack2(fg0.y, fg1.y);
      *(unsigned*)&Bg[q][bn4 + 2][bk2] = pack2(fg0.z, fg1.z);
      *(unsigned*)&Bg[q][bn4 + 3][bk2] = pack2(fg0.w, fg1.w);
      *(unsigned*)&Bu[q][bn4 + 0][bk2] = pack2(fu0.x, fu1.x);
      *(unsigned*)&Bu[q][bn4 + 1][bk2] = pack2(fu0.y, fu1.y);
      *(unsigned*)&Bu[q][bn4 + 2][bk2] = pack2(fu0.z, fu1.z);
      *(unsigned*)&Bu[q][bn4 + 3][bk2] = pack2(fu0.w, fu1.w);
#if MOE_ASYNC_LDS
      async_wait0();
#endif
      __syncthreads();
    }
  }

  // epilogue: h = silu(g) * u, bf16  (v_rcp_f32 instead of IEEE divide)
  int moff = (lane >> 4) * 8;
  int col = n0 + wn * 16 + lm;
#pragma unroll
  for (int i = 0; i < 4; ++i) {
    int rb = m0 + wm * 64 + i * 16 + moff;
#pragma unroll
    for (int r = 0; r < 8; ++r) {
      int row = rb + r;
      if (row < count) {
        float gv = accG[i][r], uv = accU[i][r];
        float sig = __builtin_amdgcn_rcpf(1.0f + __expf(-gv));
        h[(size_t)row * N + col] = (bf16_t)(gv * sig * uv);
      }
    }
  }
}

// ---------------------------------------------------------------------------
// down GEMM: h[count, INTER] @ Wd[INTER, EMBED]; scatter rows via idx scaled
// by wgt (atomic add) for routed experts, plain store for the shared expert.
// ---------------------------------------------------------------------------
__global__ __launch_bounds__(256)
void moe_down_kernel(const bf16_t* __restrict__ h,
                     const float* __restrict__ Wd,
                     float* __restrict__ out,
                     const int* __restrict__ idx,
                     const float* __restrict__ wgt,
                     const int* __restrict__ countPtr,
                     int doAtomic) {
  const int K = INTER, N = EMBED;
  __shared__ bf16_t As[2][128][40];
  __shared__ bf16_t Bs[2][64][40];
  __shared__ int   orow[128];
  __shared__ float wrow[128];

  int count = countPtr ? *countPtr : TTOT;
  int m0 = blockIdx.y * 128;
  if (m0 >= count) return;
  int n0 = blockIdx.x * 64;
  int tid = threadIdx.x;

  if (tid < 128) {
    int rc = imin(m0 + tid, count - 1);
    orow[tid] = idx ? idx[rc] : rc;
    wrow[tid] = wgt ? wgt[rc] : 1.0f;
  }
  __syncthreads();

  int lane = tid & 31, wid = tid >> 5;
  int wm = wid >> 2, wn = wid & 3;
  int lm = lane & 15;
  int lk8  = (lane >> 4) * 8;
  int lk16 = (lane >> 4) * 16;
  int bnrow = wn * 16 + lm;

  fx8 acc[4];
#pragma unroll
  for (int i = 0; i < 4; ++i) acc[i] = (fx8)0.0f;

  int ar = tid >> 1, ah = (tid & 1) * 16;
  int bn4 = (tid & 15) * 4;
  int bk2 = (tid >> 4) * 2;

  int arow = imin(m0 + ar, count - 1);   // h rows are list positions
  const bf16_t* aptr = h + (size_t)arow * K + ah;
  const float*  wptr = Wd + (size_t)bk2 * N + n0 + bn4;

#if !MOE_ASYNC_LDS
  uint4 pa0, pa1;
#endif
  float4 fw0, fw1;

#if MOE_ASYNC_LDS
  ASYNC_CP16(&As[0][ar][ah],     aptr);
  ASYNC_CP16(&As[0][ar][ah + 8], aptr + 8);
#else
  { const uint4* s = (const uint4*)aptr; pa0 = s[0]; pa1 = s[1]; }
#endif
  fw0 = *(const float4*)wptr; fw1 = *(const float4*)(wptr + N);
#if !MOE_ASYNC_LDS
  *(uint4*)&As[0][ar][ah]     = pa0;
  *(uint4*)&As[0][ar][ah + 8] = pa1;
#endif
  *(unsigned*)&Bs[0][bn4 + 0][bk2] = pack2(fw0.x, fw1.x);
  *(unsigned*)&Bs[0][bn4 + 1][bk2] = pack2(fw0.y, fw1.y);
  *(unsigned*)&Bs[0][bn4 + 2][bk2] = pack2(fw0.z, fw1.z);
  *(unsigned*)&Bs[0][bn4 + 3][bk2] = pack2(fw0.w, fw1.w);
#if MOE_ASYNC_LDS
  async_wait0();
#endif
  __syncthreads();

  const int NK = K / 32;
  for (int it = 0; it < NK; ++it) {
    int p = it & 1;
    bool more = (it + 1) < NK;
    if (more) {
      const bf16_t* ap = aptr + (it + 1) * 32;
      const float*  wp = wptr + (size_t)(it + 1) * 32 * N;
      __builtin_prefetch(wp + (size_t)32 * N, 0, 0);
#if MOE_ASYNC_LDS
      int q = p ^ 1;
      ASYNC_CP16(&As[q][ar][ah],     ap);
      ASYNC_CP16(&As[q][ar][ah + 8], ap + 8);
#else
      { const uint4* s = (const uint4*)ap; pa0 = s[0]; pa1 = s[1]; }
#endif
      fw0 = *(const float4*)wp; fw1 = *(const float4*)(wp + N);
    }

    bfx16 b = cat8(*(const bfx8*)&Bs[p][bnrow][lk8],
                   *(const bfx8*)&Bs[p][bnrow][16 + lk8]);
    bfx16 afrag[4];
#pragma unroll
    for (int i = 0; i < 4; ++i) {
      int r = wm * 64 + i * 16 + lm;
      afrag[i] = cat8(*(const bfx8*)&As[p][r][lk8],
                      *(const bfx8*)&As[p][r][16 + lk8]);
    }
#pragma unroll
    for (int i = 0; i < 4; ++i) {
      acc[i] = __builtin_amdgcn_wmma_f32_16x16x32_bf16(false, afrag[i], false, b,
                                                       (short)0, acc[i], false, false);
    }

    if (more) {
      int q = p ^ 1;
#if !MOE_ASYNC_LDS
      *(uint4*)&As[q][ar][ah]     = pa0;
      *(uint4*)&As[q][ar][ah + 8] = pa1;
#endif
      *(unsigned*)&Bs[q][bn4 + 0][bk2] = pack2(fw0.x, fw1.x);
      *(unsigned*)&Bs[q][bn4 + 1][bk2] = pack2(fw0.y, fw1.y);
      *(unsigned*)&Bs[q][bn4 + 2][bk2] = pack2(fw0.z, fw1.z);
      *(unsigned*)&Bs[q][bn4 + 3][bk2] = pack2(fw0.w, fw1.w);
#if MOE_ASYNC_LDS
      async_wait0();
#endif
      __syncthreads();
    }
  }

  int moff = (lane >> 4) * 8;
  int col = n0 + wn * 16 + lm;
#pragma unroll
  for (int i = 0; i < 4; ++i) {
    int lb = wm * 64 + i * 16 + moff;
#pragma unroll
    for (int r = 0; r < 8; ++r) {
      int lrow = lb + r;
      int row = m0 + lrow;
      if (row < count) {
        float v = acc[i][r] * wrow[lrow];
        float* dst = out + (size_t)orow[lrow] * N + col;
        if (doAtomic) atomicAdd(dst, v);
        else          *dst = v;
      }
    }
  }
}

// ---------------------------------------------------------------------------
// aux loss finalize
// ---------------------------------------------------------------------------
__global__ void moe_aux_kernel(const int* counts, const float* Psum, float* out_aux) {
  float aux = 0.0f;
  for (int e = 0; e < NEXP; ++e) {
    float f = (float)counts[e] / (float)(TTOT * TOPK);
    float P = Psum[e] / (float)TTOT;
    aux += f * P;
  }
  *out_aux = (float)NEXP * aux;
}

// ---------------------------------------------------------------------------
extern "C" void kernel_launch(void* const* d_in, const int* in_sizes, int n_in,
                              void* d_out, int out_size, void* d_ws, size_t ws_size,
                              hipStream_t stream) {
  (void)in_sizes; (void)n_in; (void)out_size; (void)ws_size;
  const float* x   = (const float*)d_in[0];
  const float* gw  = (const float*)d_in[1];
  const float* shg = (const float*)d_in[2];
  const float* shu = (const float*)d_in[3];
  const float* shd = (const float*)d_in[4];
  const float* eg  = (const float*)d_in[5];
  const float* eu  = (const float*)d_in[6];
  const float* ed  = (const float*)d_in[7];
  float* out = (float*)d_out;

  char* ws = (char*)d_ws;
  size_t off = 0;
  auto alloc = [&](size_t bytes) {
    void* p = ws + off;
    off = (off + bytes + 255) & ~(size_t)255;
    return p;
  };
  bf16_t* xb     = (bf16_t*)alloc((size_t)TTOT * EMBED * sizeof(bf16_t));
  bf16_t* h      = (bf16_t*)alloc((size_t)TTOT * INTER * sizeof(bf16_t));
  int*    idxArr = (int*)   alloc((size_t)NEXP * TTOT * sizeof(int));
  float*  wgtArr = (float*) alloc((size_t)NEXP * TTOT * sizeof(float));
  int*    counts = (int*)   alloc(256);
  float*  Psum   = (float*) alloc(256);

  moe_init_kernel<<<1, 32, 0, stream>>>(counts, Psum);
  moe_router_kernel<<<TTOT / 256, 256, 0, stream>>>(x, gw, counts, idxArr, wgtArr, Psum);
  moe_cast_kernel<<<(TTOT * EMBED) / (256 * 8), 256, 0, stream>>>(x, xb);

  dim3 gridGU(INTER / 64, TTOT / 128);
  dim3 gridDN(EMBED / 64, TTOT / 128);

  // shared expert: full-density, plain stores initialize out
  moe_gateup_kernel<<<gridGU, 256, 0, stream>>>(xb, shg, shu, h, nullptr, nullptr);
  moe_down_kernel<<<gridDN, 256, 0, stream>>>(h, shd, out, nullptr, nullptr, nullptr, 0);

  // routed experts: gather -> gateup -> down(scatter, scale, atomic add)
  for (int e = 0; e < NEXP; ++e) {
    moe_gateup_kernel<<<gridGU, 256, 0, stream>>>(
        xb, eg + (size_t)e * EMBED * INTER, eu + (size_t)e * EMBED * INTER,
        h, idxArr + (size_t)e * TTOT, counts + e);
    moe_down_kernel<<<gridDN, 256, 0, stream>>>(
        h, ed + (size_t)e * INTER * EMBED, out,
        idxArr + (size_t)e * TTOT, wgtArr + (size_t)e * TTOT, counts + e, 1);
  }

  moe_aux_kernel<<<1, 1, 0, stream>>>(counts, Psum, out + (size_t)TTOT * EMBED);
}
